// Attention_6837587935759
// MI455X (gfx1250) — compile-verified, matrix operand order checked by reading, not run on hardware
//
#include <hip/hip_runtime.h>

// ---------------------------------------------------------------------------
// MMDiT joint attention for MI455X (gfx1250, wave32, WMMA).
// fp32 in/out; bf16 WMMA (V_WMMA_F32_16X16X32_BF16) with fp32 accumulation.
// ---------------------------------------------------------------------------

typedef __bf16 bf16;
typedef __attribute__((ext_vector_type(16))) __bf16 v16bf;
typedef __attribute__((ext_vector_type(8)))  float   v8f;

union FragA { v16bf v; float4 q[2]; };   // 16x32 bf16 A/B fragment (8 VGPRs)
union FragC { v8f   v; float  f[8]; };   // 16x16 f32 C/D fragment (8 VGPRs)

#define BB   2
#define NN   1024
#define LL   2048          // N + M
#define DD   1536
#define NHH  16
#define HDD  96
#define EPSF 1e-6f
#define SCALEF 0.10206207261596575f   // 96^-0.5

// ---- gfx1250 async Global->LDS path (ASYNCcnt), with portable fallback ----
#if defined(__has_builtin)
#if __has_builtin(__builtin_amdgcn_global_load_async_to_lds_b32)
#define HAVE_ASYNC_LDS 1
#endif
#endif
#ifndef HAVE_ASYNC_LDS
#define HAVE_ASYNC_LDS 0
#endif

typedef __attribute__((address_space(1))) void as1_void;
typedef __attribute__((address_space(3))) void as3_void;
typedef __attribute__((address_space(1))) int  as1_int;
typedef __attribute__((address_space(3))) int  as3_int;

__device__ __forceinline__ void cp_lds_b32(unsigned* ldsp, const unsigned* gp) {
#if HAVE_ASYNC_LDS
  __builtin_amdgcn_global_load_async_to_lds_b32(
      (as1_int*)(as1_void*)(unsigned*)gp,
      (as3_int*)(as3_void*)ldsp, 0, 0);
#else
  *ldsp = *gp;
#endif
}

__device__ __forceinline__ void wait_async_lds() {
#if HAVE_ASYNC_LDS
#if defined(__has_builtin) && __has_builtin(__builtin_amdgcn_s_wait_asynccnt)
  __builtin_amdgcn_s_wait_asynccnt(0);
#else
  asm volatile("s_wait_asynccnt 0x0" ::: "memory");
#endif
#endif
}

__device__ __forceinline__ unsigned pack2(float a, float b) {
  union { __bf16 h[2]; unsigned u; } p;
  p.h[0] = (__bf16)a; p.h[1] = (__bf16)b;
  return p.u;
}

__device__ __forceinline__ v8f wmma_bf16(v16bf a, v16bf b, v8f c) {
  // D = A(16x32) * B(32x16) + C ; emits v_wmma_f32_16x16x32_bf16
  return __builtin_amdgcn_wmma_f32_16x16x32_bf16(false, a, false, b,
                                                 (short)0, c, false, false);
}

// ---------------------------------------------------------------------------
// Kernel 1/4: generic GEMM  C[map_out(r), :] = A[map_in(r), :] @ W
//   A: fp32, row length DD;  W: fp32 [DD x DD] row-major (k,n);  C: fp32.
//   row mapping: phys = off + (r>>10)*bs + (r&1023)   (rows == 2048)
//   Block tile 128x64, 8 waves each 32x32; K-step 64 (2 WMMA k-slices/stage).
// ---------------------------------------------------------------------------
__global__ __launch_bounds__(256)
void gemm_rowmap(const float* __restrict__ A, const float* __restrict__ W,
                 float* __restrict__ C,
                 int in_off, int in_bs, int out_off, int out_bs)
{
  __shared__ bf16 la[128][72];   // A tile, 144B rows (16B aligned)
  __shared__ bf16 lw[64][72];    // W tile transposed: lw[n][k]

  const int t    = threadIdx.x;
  const int lane = t & 31;
  const int wv   = t >> 5;
  const int wm   = wv >> 1;          // 0..3 (M direction)
  const int wn   = wv & 1;           // 0..1 (N direction)
  const int half = lane >> 4;        // K-chunk select per ISA layout
  const int lm   = lane & 15;
  const int bm   = blockIdx.x * 128;
  const int bn   = blockIdx.y * 64;

  FragC acc[2][2];
#pragma unroll
  for (int i = 0; i < 2; ++i)
#pragma unroll
    for (int j = 0; j < 2; ++j)
#pragma unroll
      for (int r = 0; r < 8; ++r) acc[i][j].f[r] = 0.f;

  // staging assignments
  const int ar  = t & 127;             // A row
  const int ac0 = (t >> 7) * 32;       // A col group (0 / 32)
  const int g_in = bm + ar;
  const int in_row = in_off + (g_in >> 10) * in_bs + (g_in & 1023);
  const float* arow = A + (size_t)in_row * DD;
  const int wcol = t & 63;             // W output column n
  const int wkg  = t >> 6;             // k group (0..3) * 16

  for (int k0 = 0; k0 < DD; k0 += 64) {
    // ---- stage A tile (fp32 -> packed bf16, b128 LDS stores)
    {
      const float4* src = (const float4*)(arow + k0 + ac0);
      if (k0 + 64 < DD) __builtin_prefetch(arow + k0 + 64 + ac0, 0, 3);
#pragma unroll
      for (int gq = 0; gq < 4; ++gq) {
        const float4 f0 = src[gq * 2], f1 = src[gq * 2 + 1];
        uint4 u;
        u.x = pack2(f0.x, f0.y); u.y = pack2(f0.z, f0.w);
        u.z = pack2(f1.x, f1.y); u.w = pack2(f1.z, f1.w);
        *(uint4*)&la[ar][ac0 + gq * 8] = u;
      }
    }
    // ---- stage W tile transposed (column gathers, b128 LDS stores)
    {
      const float* src = W + (size_t)(k0 + wkg * 16) * DD + bn + wcol;
#pragma unroll
      for (int hh = 0; hh < 2; ++hh) {
        float f[8];
#pragma unroll
        for (int i = 0; i < 8; ++i) f[i] = src[(size_t)(hh * 8 + i) * DD];
        uint4 u;
        u.x = pack2(f[0], f[1]); u.y = pack2(f[2], f[3]);
        u.z = pack2(f[4], f[5]); u.w = pack2(f[6], f[7]);
        *(uint4*)&lw[wcol][wkg * 16 + hh * 8] = u;
      }
    }
    __syncthreads();

#pragma unroll
    for (int kk = 0; kk < 2; ++kk) {
      FragA af[2], bfg[2];
#pragma unroll
      for (int i = 0; i < 2; ++i) {
        const bf16* p = &la[wm * 32 + i * 16 + lm][kk * 32 + half * 8];
        af[i].q[0] = *(const float4*)p;
        af[i].q[1] = *(const float4*)(p + 16);
      }
#pragma unroll
      for (int j = 0; j < 2; ++j) {
        const bf16* p = &lw[wn * 32 + j * 16 + lm][kk * 32 + half * 8];
        bfg[j].q[0] = *(const float4*)p;
        bfg[j].q[1] = *(const float4*)(p + 16);
      }
#pragma unroll
      for (int i = 0; i < 2; ++i)
#pragma unroll
        for (int j = 0; j < 2; ++j)
          acc[i][j].v = wmma_bf16(af[i].v, bfg[j].v, acc[i][j].v);
    }
    __syncthreads();
  }

  // ---- epilogue: C layout = lane holds column lm, rows half*8 + r
  const int m0 = half * 8;
#pragma unroll
  for (int i = 0; i < 2; ++i)
#pragma unroll
    for (int j = 0; j < 2; ++j) {
      const int gn = bn + wn * 32 + j * 16 + lm;
#pragma unroll
      for (int r = 0; r < 8; ++r) {
        const int g = bm + wm * 32 + i * 16 + m0 + r;
        const int out_row = out_off + (g >> 10) * out_bs + (g & 1023);
        C[(size_t)out_row * DD + gn] = acc[i][j].f[r];
      }
    }
}

// ---------------------------------------------------------------------------
// Kernel 2: per-token RMSNorm (over full D) + 3D RoPE + pack to bf16
//   qraw/kraw/vraw: fp32 [B][L][D] ->
//     qb/kb: bf16 [B][NH][L][HD]     (WMMA A / B row-major layouts)
//     vb   : bf16 [B][NH][HD][L]     (pre-transposed for the P*V B-matrix)
//   One block per (b, l). pass 0 = q, 1 = k (norm+rope), 2 = v (copy).
// ---------------------------------------------------------------------------
__global__ __launch_bounds__(256)
void norm_rope_pack(const float* __restrict__ qraw, const float* __restrict__ kraw,
                    const float* __restrict__ vraw,
                    const float* __restrict__ c_qn, const float* __restrict__ c_kn,
                    const float* __restrict__ x_qn, const float* __restrict__ x_kn,
                    bf16* __restrict__ qb, bf16* __restrict__ kb, bf16* __restrict__ vb)
{
  __shared__ float row[DD];
  __shared__ float red[256];

  const int l = blockIdx.x, b = blockIdx.y, t = threadIdx.x;
  const float pt = (float)(l >> 10);          // t index (T=2,H=32,W=32)
  const float ph = (float)((l >> 5) & 31);
  const float pw = (float)(l & 31);
  const bool isCond = (l < NN);
  const size_t base = ((size_t)b * LL + l) * DD;

  for (int pass = 0; pass < 3; ++pass) {
    const float* src = (pass == 0) ? qraw : (pass == 1) ? kraw : vraw;
    bf16* dst        = (pass == 0) ? qb   : (pass == 1) ? kb   : vb;
    const float* wn  = (pass == 0) ? (isCond ? c_qn : x_qn)
                                   : (isCond ? c_kn : x_kn);

    float v[6]; float ss = 0.f;
#pragma unroll
    for (int i = 0; i < 6; ++i) {
      v[i] = src[base + t + i * 256];
      ss += v[i] * v[i];
    }
    float rms = 1.f;
    if (pass < 2) {
      red[t] = ss;
      __syncthreads();
      for (int s = 128; s > 0; s >>= 1) {
        if (t < s) red[t] += red[t + s];
        __syncthreads();
      }
      rms = rsqrtf(red[0] / (float)DD + EPSF);
    }
#pragma unroll
    for (int i = 0; i < 6; ++i) {
      const int d = t + i * 256;
      row[d] = (pass < 2) ? v[i] * rms * wn[d] : v[i];
    }
    __syncthreads();
#pragma unroll
    for (int i = 0; i < 6; ++i) {
      const int d = t + i * 256;
      const int h = d / HDD, e = d % HDD;
      float outv = row[d];
      size_t di;
      if (pass < 2) {
        const int seg = e / 32, j = e & 31, fi = j & 15;
        const float pos  = (seg == 0) ? pt : (seg == 1) ? ph : pw;
        // inv_freq = 10000^(-fi/16)
        const float ang  = pos * __expf(-(float)fi * (9.210340371976184f / 16.f));
        const float c = __cosf(ang), s = __sinf(ang);
        const float partner = (j < 16) ? row[d + 16] : row[d - 16];
        const float rot     = (j < 16) ? -partner : partner;
        outv = outv * c + rot * s;
        di = (((size_t)b * NHH + h) * LL + l) * HDD + e;          // [B][NH][L][HD]
      } else {
        di = (((size_t)b * NHH + h) * HDD + e) * LL + l;          // [B][NH][HD][L]
      }
      dst[di] = (bf16)outv;
    }
    __syncthreads();
  }
}

// ---------------------------------------------------------------------------
// Kernel 3: flash attention (online softmax) with WMMA.
//   Block = (128 queries) x head x batch; 8 waves, each owns 16 queries.
//   Loop over 32-key tiles: S = Q*K^T (6 wmma), P*V into O (6 wmma).
//   K/V tiles staged via async Global->LDS dword copies (ASYNCcnt) when
//   available; V arrives pre-transposed so staging is pure row copies.
// ---------------------------------------------------------------------------
__global__ __launch_bounds__(256)
void flash_attn(const bf16* __restrict__ qb, const bf16* __restrict__ kb,
                const bf16* __restrict__ vb, float* __restrict__ aout)
{
  __shared__ bf16 lk[32][104];      // K tile row-major, 208B rows (16B aligned)
  __shared__ bf16 lv[96][40];       // V tile transposed: lv[hd][key], 80B rows
  __shared__ bf16 lp[8][16][32];    // per-wave P staging (C-layout -> A-layout)

  const int t    = threadIdx.x;
  const int lane = t & 31;
  const int wv   = t >> 5;
  const int half = lane >> 4;
  const int lm   = lane & 15;
  const int qblk = blockIdx.x * 128;
  const int h    = blockIdx.y;
  const int b    = blockIdx.z;
  const size_t headbase  = ((size_t)b * NHH + h) * LL * HDD;   // qb / kb
  const size_t headbaseT = ((size_t)b * NHH + h) * HDD * LL;   // vb (transposed)

  // Q fragments resident for the whole kernel (A layout: lane = row lm)
  FragA qf[3];
  {
    const bf16* qp = qb + headbase + (size_t)(qblk + wv * 16 + lm) * HDD;
#pragma unroll
    for (int kk = 0; kk < 3; ++kk) {
      const bf16* p = qp + kk * 32 + half * 8;
      qf[kk].q[0] = *(const float4*)p;
      qf[kk].q[1] = *(const float4*)(p + 16);
    }
  }

  FragC o[6];
#pragma unroll
  for (int f6 = 0; f6 < 6; ++f6)
#pragma unroll
    for (int r = 0; r < 8; ++r) o[f6].f[r] = 0.f;
  float rmax[8], rsum[8];
#pragma unroll
  for (int r = 0; r < 8; ++r) { rmax[r] = -3.0e38f; rsum[r] = 0.f; }

  for (int kt = 0; kt < LL; kt += 32) {
    // ---- stage K tile: 32 rows x 48 dwords
    {
      const unsigned* src = (const unsigned*)(kb + headbase + (size_t)kt * HDD);
      if (kt + 32 < LL)
        __builtin_prefetch((const char*)(src + 32 * 48) + t * 16, 0, 3);
#pragma unroll
      for (int i = 0; i < 6; ++i) {
        const int f = t + i * 256;
        const int r = f / 48, c = f % 48;
        cp_lds_b32((unsigned*)&lk[r][0] + c, src + r * 48 + c);
      }
    }
    // ---- stage V tile (already transposed): 96 rows x 16 dwords
    {
      const unsigned* src = (const unsigned*)(vb + headbaseT + kt);
#pragma unroll
      for (int i = 0; i < 6; ++i) {
        const int f = t + i * 256;
        const int hd = f >> 4, c = f & 15;
        cp_lds_b32((unsigned*)&lv[hd][0] + c, src + hd * (LL / 2) + c);
      }
    }
    wait_async_lds();
    __syncthreads();

    // ---- S = Q * K^T  (two 16-key column fragments, 3 k-steps over HD)
    FragC s[2];
#pragma unroll
    for (int n16 = 0; n16 < 2; ++n16) {
#pragma unroll
      for (int r = 0; r < 8; ++r) s[n16].f[r] = 0.f;
#pragma unroll
      for (int kk = 0; kk < 3; ++kk) {
        FragA kf;
        const bf16* p = &lk[n16 * 16 + lm][kk * 32 + half * 8];
        kf.q[0] = *(const float4*)p;
        kf.q[1] = *(const float4*)(p + 16);
        s[n16].v = wmma_bf16(qf[kk].v, kf.v, s[n16].v);
      }
    }

    // ---- online softmax (row stats across 16-lane halves)
    float tmax[8];
#pragma unroll
    for (int r = 0; r < 8; ++r) {
      s[0].f[r] *= SCALEF; s[1].f[r] *= SCALEF;
      tmax[r] = fmaxf(s[0].f[r], s[1].f[r]);
    }
#pragma unroll
    for (int off = 1; off < 16; off <<= 1)
#pragma unroll
      for (int r = 0; r < 8; ++r)
        tmax[r] = fmaxf(tmax[r], __shfl_xor(tmax[r], off));

    float fac[8], tsum[8];
#pragma unroll
    for (int r = 0; r < 8; ++r) {
      const float mn = fmaxf(rmax[r], tmax[r]);
      fac[r] = __expf(rmax[r] - mn);
      rmax[r] = mn;
      s[0].f[r] = __expf(s[0].f[r] - mn);
      s[1].f[r] = __expf(s[1].f[r] - mn);
      tsum[r] = s[0].f[r] + s[1].f[r];
    }
#pragma unroll
    for (int off = 1; off < 16; off <<= 1)
#pragma unroll
      for (int r = 0; r < 8; ++r) tsum[r] += __shfl_xor(tsum[r], off);
#pragma unroll
    for (int r = 0; r < 8; ++r) rsum[r] = rsum[r] * fac[r] + tsum[r];
#pragma unroll
    for (int f6 = 0; f6 < 6; ++f6)
#pragma unroll
      for (int r = 0; r < 8; ++r) o[f6].f[r] *= fac[r];

    // ---- P: C-layout -> bf16 in LDS -> reload as A fragment (wave-local)
#pragma unroll
    for (int n16 = 0; n16 < 2; ++n16)
#pragma unroll
      for (int r = 0; r < 8; ++r)
        lp[wv][half * 8 + r][n16 * 16 + lm] = (bf16)s[n16].f[r];

    FragA pf;
    {
      const bf16* p = &lp[wv][lm][half * 8];
      pf.q[0] = *(const float4*)p;
      pf.q[1] = *(const float4*)(p + 16);
    }
    // ---- O += P(16x32) * V(32x96)
#pragma unroll
    for (int f6 = 0; f6 < 6; ++f6) {
      FragA vf;
      const bf16* p = &lv[f6 * 16 + lm][half * 8];
      vf.q[0] = *(const float4*)p;
      vf.q[1] = *(const float4*)(p + 16);
      o[f6].v = wmma_bf16(pf.v, vf.v, o[f6].v);
    }
    __syncthreads();
  }

  // ---- finalize: O /= l ; write fp32 [B][L][D]
  float rinv[8];
#pragma unroll
  for (int r = 0; r < 8; ++r) rinv[r] = 1.f / rsum[r];
  const int qrow0 = qblk + wv * 16 + half * 8;
#pragma unroll
  for (int f6 = 0; f6 < 6; ++f6)
#pragma unroll
    for (int r = 0; r < 8; ++r)
      aout[((size_t)b * LL + qrow0 + r) * DD + h * HDD + f6 * 16 + lm] =
          o[f6].f[r] * rinv[r];
}

// ---------------------------------------------------------------------------
// Launch
// ---------------------------------------------------------------------------
extern "C" void kernel_launch(void* const* d_in, const int* in_sizes, int n_in,
                              void* d_out, int out_size, void* d_ws, size_t ws_size,
                              hipStream_t stream) {
  const float* cond      = (const float*)d_in[0];
  const float* x         = (const float*)d_in[1];
  const float* cond_q_w  = (const float*)d_in[2];
  const float* cond_k_w  = (const float*)d_in[3];
  const float* cond_v_w  = (const float*)d_in[4];
  const float* cond_qn   = (const float*)d_in[5];
  const float* cond_kn   = (const float*)d_in[6];
  const float* cond_p_w  = (const float*)d_in[7];
  const float* x_q_w     = (const float*)d_in[8];
  const float* x_k_w     = (const float*)d_in[9];
  const float* x_v_w     = (const float*)d_in[10];
  const float* x_qn      = (const float*)d_in[11];
  const float* x_kn      = (const float*)d_in[12];
  const float* x_p_w     = (const float*)d_in[13];
  // d_in[14..16] = T,H,W (2,32,32) — baked into the RoPE index math.

  const size_t ROWSxD = (size_t)BB * LL * DD;          // 6,291,456
  const size_t HEADEL = (size_t)BB * NHH * LL * HDD;   // 6,291,456
  float* qraw = (float*)d_ws;
  float* kraw = qraw + ROWSxD;
  float* vraw = kraw + ROWSxD;
  bf16*  qb   = (bf16*)(vraw + ROWSxD);
  bf16*  kb   = qb + HEADEL;
  bf16*  vb   = kb + HEADEL;
  float* aout = qraw;   // alias: raw Q dead after norm_rope_pack

  const dim3 gg(16, 24);   // 2048/128 x 1536/64
  // QKV GEMMs: cond rows -> tokens [0,1024), x rows -> tokens [1024,2048)
  gemm_rowmap<<<gg, 256, 0, stream>>>(cond, cond_q_w, qraw, 0, 1024, 0,    2048);
  gemm_rowmap<<<gg, 256, 0, stream>>>(cond, cond_k_w, kraw, 0, 1024, 0,    2048);
  gemm_rowmap<<<gg, 256, 0, stream>>>(cond, cond_v_w, vraw, 0, 1024, 0,    2048);
  gemm_rowmap<<<gg, 256, 0, stream>>>(x,    x_q_w,    qraw, 0, 1024, 1024, 2048);
  gemm_rowmap<<<gg, 256, 0, stream>>>(x,    x_k_w,    kraw, 0, 1024, 1024, 2048);
  gemm_rowmap<<<gg, 256, 0, stream>>>(x,    x_v_w,    vraw, 0, 1024, 1024, 2048);

  norm_rope_pack<<<dim3(LL, BB), 256, 0, stream>>>(
      qraw, kraw, vraw, cond_qn, cond_kn, x_qn, x_kn, qb, kb, vb);

  flash_attn<<<dim3(LL / 128, NHH, BB), 256, 0, stream>>>(qb, kb, vb, aout);

  // Projections: cond_out then x_out, concatenated in d_out
  float* out_c = (float*)d_out;
  float* out_x = out_c + (size_t)BB * NN * DD;
  gemm_rowmap<<<gg, 256, 0, stream>>>(aout, cond_p_w, out_c, 0,    2048, 0, 1024);
  gemm_rowmap<<<gg, 256, 0, stream>>>(aout, x_p_w,    out_x, 1024, 2048, 0, 1024);
}
